// Decoder_79087527788964
// MI455X (gfx1250) — matrix operand order repaired and checked by reading, not compiled, not run on hardware
//
#include <hip/hip_runtime.h>
#include <hip/hip_bf16.h>

// Sizes
#define B_    32
#define T_    800
#define NMEL  80
#define ATTN_ 512
#define PRE_  256
#define RNN_  1024
#define DIN   768            // PRE_ + ATTN_
#define KTOT  (DIN + RNN_)   // 1792
#define ROWS  (B_ * T_)      // 25600
#define NGATE (4 * RNN_)     // 4096
#define NWG   32             // persistent LSTM workgroups
#define ALDS  1800           // padded LDS row stride (ushorts): 3600B = 900 dwords, 900%64=4

typedef __bf16 bf16_t;
typedef bf16_t bf16x8  __attribute__((ext_vector_type(8)));
typedef bf16_t bf16x16 __attribute__((ext_vector_type(16)));
typedef float  f32x8   __attribute__((ext_vector_type(8)));

static __device__ __forceinline__ unsigned short f2bf(float f) {
  unsigned u = __builtin_bit_cast(unsigned, f);
  u += 0x7FFFu + ((u >> 16) & 1u);           // round-to-nearest-even
  return (unsigned short)(u >> 16);
}

static __device__ __forceinline__ bf16x16 joinv(bf16x8 lo, bf16x8 hi) {
  return __builtin_shufflevector(lo, hi, 0,1,2,3,4,5,6,7,8,9,10,11,12,13,14,15);
}
// A fragment for 16x16x32 bf16 WMMA: per lane, elems 0..7 at p[0..7], elems 8..15 at p[16..23]
static __device__ __forceinline__ bf16x16 ldA(const unsigned short* p) {
  bf16x8 lo = *(const bf16x8*)(p);
  bf16x8 hi = *(const bf16x8*)(p + 16);
  return joinv(lo, hi);
}
// B fragment: per lane, elems 0..7 at p[0..7], elems 8..15 at p[16..23]
static __device__ __forceinline__ bf16x16 ldB(const unsigned short* p) {
  bf16x8 lo = *(const bf16x8*)(p);
  bf16x8 hi = *(const bf16x8*)(p + 16);
  return joinv(lo, hi);
}
static __device__ __forceinline__ f32x8 wmma_bf16(bf16x16 a, bf16x16 b, f32x8 c) {
  return __builtin_amdgcn_wmma_f32_16x16x32_bf16(false, a, false, b, (short)0, c, false, false);
}
static __device__ __forceinline__ float sigf(float x) {
  return 1.0f / (1.0f + __expf(-x));
}

// Device-scope sense-reversing grid barrier (32 co-resident WGs)
static __device__ __forceinline__ void grid_barrier(unsigned* bar) {
  __threadfence();
  __syncthreads();
  if (threadIdx.x == 0) {
    unsigned ph  = __hip_atomic_load(&bar[1], __ATOMIC_RELAXED, __HIP_MEMORY_SCOPE_AGENT);
    unsigned arr = __hip_atomic_fetch_add(&bar[0], 1u, __ATOMIC_ACQ_REL, __HIP_MEMORY_SCOPE_AGENT);
    if (arr == NWG - 1) {
      __hip_atomic_store(&bar[0], 0u, __ATOMIC_RELAXED, __HIP_MEMORY_SCOPE_AGENT);
      __hip_atomic_store(&bar[1], ph + 1u, __ATOMIC_RELEASE, __HIP_MEMORY_SCOPE_AGENT);
    } else {
      while (__hip_atomic_load(&bar[1], __ATOMIC_ACQUIRE, __HIP_MEMORY_SCOPE_AGENT) == ph)
        __builtin_amdgcn_s_sleep(1);
    }
  }
  __syncthreads();
}

// ---------------- Kernel 1: weight conversion + K-concat + buffer init ----------------
// wcat[n][k] (bf16, K=1792) = [ W_ih[n][0:768] | W_hh[n][0:1024] ]  -> one linear B pointer.
__global__ __launch_bounds__(256) void k_prep(
    const float* __restrict__ w_ih, const float* __restrict__ w_hh,
    const float* __restrict__ proj_w,
    unsigned short* __restrict__ wcat_bf,
    unsigned short* __restrict__ pw_bf, unsigned short* __restrict__ zbuf,
    unsigned* __restrict__ bar)
{
  const int stride = gridDim.x * blockDim.x;
  const int tid = blockIdx.x * blockDim.x + threadIdx.x;
  for (int i = tid; i < NGATE * DIN; i += stride) {
    const int n = i / DIN, k = i - n * DIN;
    wcat_bf[(size_t)n * KTOT + k] = f2bf(w_ih[i]);
  }
  for (int i = tid; i < NGATE * RNN_; i += stride) {
    const int n = i / RNN_, k = i - n * RNN_;
    wcat_bf[(size_t)n * KTOT + DIN + k] = f2bf(w_hh[i]);
  }
  for (int i = tid; i < NMEL * (RNN_ + ATTN_); i += stride) pw_bf[i] = f2bf(proj_w[i]);
  for (int i = tid; i < B_ * RNN_; i += stride) zbuf[i] = 0;  // bf16 zero == 0x0000
  if (tid == 0) { bar[0] = 0u; bar[1] = 0u; }
}

// ---------------- Kernel 2: prenet + build x_bf = [relu2(prenet(prev)) | memory] in bf16 ----
// Row r = t*32 + b (time-major). 8 rows per block (same t) -> each W1/W2 load feeds 8 FMAs.
#define PR 8
__global__ __launch_bounds__(256) void k_prenet(
    const float* __restrict__ memory, const float* __restrict__ y_mels,
    const float* __restrict__ w1, const float* __restrict__ w2,
    unsigned short* __restrict__ x_bf)
{
  __shared__ float sprev[PR * NMEL];   // [row][mel]
  __shared__ float sh1[PR * PRE_];     // [row][p]
  const int r0  = blockIdx.x * PR;     // 8 consecutive rows share t (8 | 32)
  const int t   = r0 >> 5;
  const int b0  = r0 & 31;
  const int tid = threadIdx.x;

  for (int idx = tid; idx < PR * NMEL; idx += 256) {
    const int rr = idx / NMEL, m = idx - rr * NMEL;
    sprev[idx] = (t == 0) ? 0.0f
               : y_mels[(size_t)(b0 + rr) * T_ * NMEL + (size_t)(t - 1) * NMEL + m];
  }
  __syncthreads();

  float acc1[PR];
  #pragma unroll
  for (int rr = 0; rr < PR; ++rr) acc1[rr] = 0.f;
  const float* w1r = w1 + tid * NMEL;
  for (int m = 0; m < NMEL; ++m) {
    const float w = w1r[m];
    #pragma unroll
    for (int rr = 0; rr < PR; ++rr) acc1[rr] += w * sprev[rr * NMEL + m];
  }
  #pragma unroll
  for (int rr = 0; rr < PR; ++rr) sh1[rr * PRE_ + tid] = fmaxf(acc1[rr], 0.f);
  __syncthreads();

  float acc2[PR];
  #pragma unroll
  for (int rr = 0; rr < PR; ++rr) acc2[rr] = 0.f;
  const float* w2r = w2 + tid * PRE_;
  for (int p = 0; p < PRE_; ++p) {
    const float w = w2r[p];
    #pragma unroll
    for (int rr = 0; rr < PR; ++rr) acc2[rr] += w * sh1[rr * PRE_ + p];
  }
  #pragma unroll
  for (int rr = 0; rr < PR; ++rr)
    x_bf[(size_t)(r0 + rr) * DIN + tid] = f2bf(fmaxf(acc2[rr], 0.f));

  for (int idx = tid; idx < PR * ATTN_; idx += 256) {
    const int rr = idx >> 9, k = idx & (ATTN_ - 1);
    x_bf[(size_t)(r0 + rr) * DIN + PRE_ + k] =
        f2bf(memory[(size_t)(b0 + rr) * T_ * ATTN_ + (size_t)t * ATTN_ + k]);
  }
}

// ---------------- Kernel 3: persistent LSTM scan ----------------
// 32 WGs x 256 threads. WG owns 32 hidden units -> 128 gate columns (i/f/g/o slices).
// Per step: stage A = [x[t] | h[t-1]] (32x1792 bf16) in LDS once, then one unified
// K loop (56 x k32) of WMMAs; simple body (no manual pipelining) so the compiler's
// unroll-4 schedule overlaps loads/WMMAs without register spills. c in registers.
__global__ __launch_bounds__(256) void k_lstm(
    const unsigned short* __restrict__ x_bf,
    const unsigned short* __restrict__ wcat_bf,
    const float* __restrict__ b_ih, const float* __restrict__ b_hh,
    unsigned short* __restrict__ hs_bf,
    const unsigned short* __restrict__ zbuf,
    unsigned* __restrict__ bar)
{
  extern __shared__ char smem[];
  unsigned short* a_lds = (unsigned short*)smem;            // 32 x ALDS ushorts = 115200 B
  float* gl = (float*)(smem + (size_t)B_ * ALDS * 2);       // 32 x 128 f32 = 16384 B

  const int tid  = threadIdx.x;
  const int wg   = blockIdx.x;
  const int wave = tid >> 5;
  const int lane = tid & 31;
  const int l16  = lane & 15;
  const int lhi  = lane >> 4;         // 0 = lanes 0-15, 1 = lanes 16-31
  const int j0   = wg * 32;           // first hidden unit owned by this WG
  const int g    = wave >> 1;         // gate 0..3 (i,f,g,o)
  const int half = wave & 1;          // 16-column half of the 32 hidden units
  const int ncol = g * RNN_ + j0 + half * 16 + l16;   // global gate column
  const int bko  = lhi * 8;           // B-fragment K sub-offset
  const int ako  = lhi * 8;           // A-fragment K sub-offset
  const unsigned short* bp = wcat_bf + (size_t)ncol * KTOT + bko;
  const int lcol = g * 32 + half * 16 + l16;
  // A-fragment LDS row bases (batch rows l16 and 16+l16)
  const unsigned short* la0 = a_lds + (size_t)l16 * ALDS + ako;
  const unsigned short* la1 = a_lds + (size_t)(16 + l16) * ALDS + ako;

  // element-wise phase mapping: thread owns hidden j, 4 batch rows
  const int jloc = tid & 31;
  const int j    = j0 + jloc;
  const int bq   = tid >> 5;
  const float bi  = b_ih[j]            + b_hh[j];
  const float bff = b_ih[RNN_ + j]     + b_hh[RNN_ + j];
  const float bgg = b_ih[2 * RNN_ + j] + b_hh[2 * RNN_ + j];
  const float boo = b_ih[3 * RNN_ + j] + b_hh[3 * RNN_ + j];
  float c[4] = {0.f, 0.f, 0.f, 0.f};

  for (int t = 0; t < T_; ++t) {
    // ---- stage A = [x[t] | h[t-1]] into LDS (cooperative, uint4 chunks) ----
    const uint4* xsrc = (const uint4*)(x_bf + (size_t)(t * B_) * DIN);  // 3072 vec16
    const uint4* hsrc = (const uint4*)((t == 0) ? zbuf
                                                : (hs_bf + (size_t)(t - 1) * B_ * RNN_)); // 4096 vec16
    for (int idx = tid; idx < B_ * DIN / 8; idx += 256) {
      const int bb = idx / (DIN / 8), v = idx - bb * (DIN / 8);
      *(uint4*)(a_lds + (size_t)bb * ALDS + v * 8) = xsrc[idx];
    }
    for (int idx = tid; idx < B_ * RNN_ / 8; idx += 256) {
      const int bb = idx >> 7, v = idx & 127;
      *(uint4*)(a_lds + (size_t)bb * ALDS + DIN + v * 8) = hsrc[idx];
    }
    __syncthreads();

    // prefetch next step's x tile into near caches while we compute (global_prefetch_b8)
    if (t + 1 < T_) {
      const char* xnext = (const char*)(x_bf + (size_t)((t + 1) * B_) * DIN);
      __builtin_prefetch(xnext + tid * 192, 0, 3);
      __builtin_prefetch(xnext + tid * 192 + 128, 0, 3);
    }

    // ---- unified K loop (56 steps of 32) ----
    f32x8 acc0 = {};   // batches 0-15
    f32x8 acc1 = {};   // batches 16-31
    #pragma unroll 4
    for (int kb = 0; kb < KTOT / 32; ++kb) {
      const int k0 = kb * 32;
      bf16x16 bfrag = ldB(bp + k0);
      acc0 = wmma_bf16(ldA(la0 + k0), bfrag, acc0);
      acc1 = wmma_bf16(ldA(la1 + k0), bfrag, acc1);
    }

    #pragma unroll
    for (int i = 0; i < 8; ++i) {        // C layout: VGPR i -> row i (+8 hi lanes)
      const int rm = i + lhi * 8;
      gl[rm * 128 + lcol]        = acc0[i];
      gl[(16 + rm) * 128 + lcol] = acc1[i];
    }
    __syncthreads();

    unsigned short* hrow = hs_bf + (size_t)t * B_ * RNN_;
    #pragma unroll
    for (int q = 0; q < 4; ++q) {
      const int bb = bq + q * 8;
      const float* grow = gl + bb * 128;
      float ig = sigf(grow[jloc]      + bi);
      float fg = sigf(grow[32 + jloc] + bff);
      float gg = tanhf(grow[64 + jloc] + bgg);
      float og = sigf(grow[96 + jloc] + boo);
      c[q] = fg * c[q] + ig * gg;
      hrow[(size_t)bb * RNN_ + j] = f2bf(og * tanhf(c[q]));
    }
    grid_barrier(bar);   // publish h[t] before any WG starts step t+1
  }
}

// ---------------- Kernel 4: projection  mel = [hs | memory] * proj_w^T + proj_b -------
// M = 25600 rows (r = t*32+b), N = 80 (5 n-tiles), K = 1536 (1024 hs + 512 memory).
__global__ __launch_bounds__(256) void k_proj(
    const unsigned short* __restrict__ hs_bf,
    const unsigned short* __restrict__ x_bf,
    const unsigned short* __restrict__ pw_bf,
    const float* __restrict__ proj_b,
    float* __restrict__ out)
{
  const int tid  = threadIdx.x;
  const int wave = tid >> 5;
  const int lane = tid & 31;
  const int l16  = lane & 15;
  const int lhi  = lane >> 4;
  const int mt   = blockIdx.x * 8 + wave;            // m-tile 0..1599
  if (mt >= ROWS / 16) return;
  const int ako = lhi * 8;
  const int bko = lhi * 8;
  const int KP  = RNN_ + ATTN_;                      // 1536
  f32x8 acc[5] = {{}, {}, {}, {}, {}};
  const unsigned short* ha = hs_bf + (size_t)(mt * 16 + l16) * RNN_ + ako;
  const unsigned short* xa = x_bf  + (size_t)(mt * 16 + l16) * DIN + PRE_ + ako;
  #pragma unroll 2
  for (int kb = 0; kb < RNN_ / 32; ++kb) {           // hs part (K 0..1023)
    const int k0 = kb * 32;
    bf16x16 a = ldA(ha + k0);
    #pragma unroll
    for (int nt = 0; nt < 5; ++nt)
      acc[nt] = wmma_bf16(a, ldB(pw_bf + (size_t)(nt * 16 + l16) * KP + bko + k0), acc[nt]);
  }
  #pragma unroll 2
  for (int kb = 0; kb < ATTN_ / 32; ++kb) {          // memory part (K 1024..1535)
    const int k0 = kb * 32;
    bf16x16 a = ldA(xa + k0);
    #pragma unroll
    for (int nt = 0; nt < 5; ++nt)
      acc[nt] = wmma_bf16(a, ldB(pw_bf + (size_t)(nt * 16 + l16) * KP + RNN_ + bko + k0), acc[nt]);
  }
  #pragma unroll
  for (int nt = 0; nt < 5; ++nt) {
    const int m  = nt * 16 + l16;
    const float pb = proj_b[m];
    #pragma unroll
    for (int i = 0; i < 8; ++i) {
      const int r = mt * 16 + i + lhi * 8;
      const int t = r >> 5;
      const int b = r & 31;
      out[(size_t)b * T_ * NMEL + (size_t)t * NMEL + m] = acc[nt][i] + pb;
    }
  }
}

// ---------------- Workspace layout (bytes, all 16B-aligned) ----------------
constexpr size_t SZ_XBF  = (size_t)ROWS * DIN * 2;              // 39,321,600
constexpr size_t SZ_WCAT = (size_t)NGATE * KTOT * 2;            // 14,680,064
constexpr size_t SZ_PW   = (size_t)NMEL * (RNN_ + ATTN_) * 2;   //    245,760
constexpr size_t SZ_HS   = (size_t)ROWS * RNN_ * 2;             // 52,428,800
constexpr size_t SZ_ZB   = (size_t)B_ * RNN_ * 2;               //     65,536
constexpr size_t OFF_XBF  = 0;
constexpr size_t OFF_WCAT = OFF_XBF  + SZ_XBF;
constexpr size_t OFF_PW   = OFF_WCAT + SZ_WCAT;
constexpr size_t OFF_HS   = OFF_PW   + SZ_PW;
constexpr size_t OFF_ZB   = OFF_HS   + SZ_HS;
constexpr size_t OFF_BAR  = OFF_ZB   + SZ_ZB;

constexpr unsigned LSTM_LDS_BYTES = (unsigned)(B_ * ALDS * 2 + B_ * 128 * 4);  // 131,584

extern "C" void kernel_launch(void* const* d_in, const int* in_sizes, int n_in,
                              void* d_out, int out_size, void* d_ws, size_t ws_size,
                              hipStream_t stream) {
  (void)in_sizes; (void)n_in; (void)out_size; (void)ws_size;
  const float* memory = (const float*)d_in[0];
  const float* y_mels = (const float*)d_in[1];
  const float* w1     = (const float*)d_in[2];
  const float* w2     = (const float*)d_in[3];
  const float* w_ih   = (const float*)d_in[4];
  const float* w_hh   = (const float*)d_in[5];
  const float* b_ih   = (const float*)d_in[6];
  const float* b_hh   = (const float*)d_in[7];
  const float* proj_w = (const float*)d_in[8];
  const float* proj_b = (const float*)d_in[9];
  float* out = (float*)d_out;

  char* ws = (char*)d_ws;
  unsigned short* x_bf    = (unsigned short*)(ws + OFF_XBF);
  unsigned short* wcat_bf = (unsigned short*)(ws + OFF_WCAT);
  unsigned short* pw_bf   = (unsigned short*)(ws + OFF_PW);
  unsigned short* hs_bf   = (unsigned short*)(ws + OFF_HS);
  unsigned short* zbuf    = (unsigned short*)(ws + OFF_ZB);
  unsigned*       bar     = (unsigned*)(ws + OFF_BAR);

  k_prep<<<1024, 256, 0, stream>>>(w_ih, w_hh, proj_w, wcat_bf, pw_bf, zbuf, bar);
  k_prenet<<<ROWS / PR, 256, 0, stream>>>(memory, y_mels, w1, w2, x_bf);
  k_lstm<<<NWG, 256, LSTM_LDS_BYTES, stream>>>(x_bf, wcat_bf, b_ih, b_hh,
                                               hs_bf, zbuf, bar);
  k_proj<<<ROWS / 16 / 8, 256, 0, stream>>>(hs_bf, x_bf, pw_bf, proj_b, out);
}
